// spiking_coESN_60696477827274
// MI455X (gfx1250) — matrix operand order repaired and checked by reading, not compiled, not run on hardware
//
#include <hip/hip_runtime.h>
#include <hip/hip_bf16.h>
#include <stdint.h>

typedef __bf16 bf16_t;
typedef __attribute__((ext_vector_type(16))) __bf16 v16bf;
typedef __attribute__((ext_vector_type(8)))  float  v8f;

#define NHID    1024
#define NBAT    128
#define TSTEPS  500
#define KSTEPS  32              // 1024 / 32 (bf16 WMMA K=32)
#define WAVES_PER_BLOCK 8
#define MAIN_BLOCKS 64          // nt == blockIdx.x, mt == wave
#define BSTRIP_ELEMS (KSTEPS * 32 * 16)   // 16384 bf16 = 32 KB per N-strip

// ---- workspace layout (bytes) ----
#define H2HP_OFF   0
#define H2HP_BYTES (NHID * NHID * 2)            // packed bf16 h2h fragments: 2 MB
#define HYB0_OFF   (H2HP_OFF + H2HP_BYTES)
#define HYB_BYTES  (NBAT * NHID * 2)            // 256 KB per buffer
#define HYB1_OFF   (HYB0_OFF + HYB_BYTES)
#define TOT_OFF    (HYB1_OFF + HYB_BYTES)       // float spike total
#define CNT_OFF    (TOT_OFF + 4)                // barrier arrive counter
#define GEN_OFF    (TOT_OFF + 8)                // barrier generation
#define ZERO_BYTES (2 * HYB_BYTES + 12)

union Frag16 { v16bf v; uint4 q[2]; };

// ---------------- grid-wide barrier (all 64 blocks resident) ----------------
__device__ __forceinline__ void grid_sync(unsigned* cnt, unsigned* gen, unsigned nblk) {
    __syncthreads();
    if (threadIdx.x == 0) {
        volatile unsigned* vg = gen;
        unsigned old = *vg;
        __threadfence();
        unsigned a = atomicAdd(cnt, 1u);
        if (a == nblk - 1u) {
            atomicExch(cnt, 0u);
            __threadfence();
            atomicAdd(gen, 1u);
        } else {
            while (*vg == old) { __builtin_amdgcn_s_sleep(2); }
        }
        __threadfence();
    }
    __syncthreads();
}

// ---------------- ws zero-init ----------------
__global__ void esn_init_kernel(unsigned* z, int nwords) {
    int i = blockIdx.x * blockDim.x + threadIdx.x;
    if (i < nwords) z[i] = 0u;
}

// ---------------- pack h2h (f32 row-major) -> bf16 B-fragment-major ----------------
// fragment (nt, ks): 32 lanes x 16 bf16, lane l element e holds
//   B[K = ks*32 + (l>>4)*16 + e][N = nt*16 + (l&15)]
__global__ void esn_pack_kernel(const float* __restrict__ h2h, bf16_t* __restrict__ h2hp) {
    int i = blockIdx.x * blockDim.x + threadIdx.x;   // 0 .. 1048575
    int e    =  i        & 15;
    int lane = (i >> 4)  & 31;
    int ks   = (i >> 9)  & 31;
    int nt   =  i >> 14;
    int K = ks * 32 + (lane >> 4) * 16 + e;
    int N = nt * 16 + (lane & 15);
    h2hp[i] = (bf16_t)h2h[K * NHID + N];
}

// branch-free tanh: 1 - 2/(exp(2z)+1); exp->inf => 1, exp->0 => -1
__device__ __forceinline__ float fast_tanh(float z) {
    float e = __expf(2.0f * z);
    return 1.0f - 2.0f * __builtin_amdgcn_rcpf(e + 1.0f);
}

// ---------------- persistent scan kernel ----------------
__global__ __launch_bounds__(256, 1) void esn_scan_kernel(
    const float* __restrict__ x,    const float* __restrict__ x2h,
    const float* __restrict__ bias, const float* __restrict__ gam,
    const float* __restrict__ eps,
    const bf16_t* __restrict__ h2hp,
    bf16_t* __restrict__ hyb0, bf16_t* __restrict__ hyb1,
    float* __restrict__ tot, unsigned* cnt, unsigned* gen,
    float* __restrict__ out)
{
    __shared__ __align__(16) bf16_t Bsh[BSTRIP_ELEMS];   // 32 KB: this block's N-strip of h2h

    const int lane = threadIdx.x & 31;
    const int wave = threadIdx.x >> 5;
    const int nt   = blockIdx.x;      // all 8 waves share the N-strip
    const int mt   = wave;            // 8 M-tiles (batch)
    const int mbase = mt << 4;
    const int l16  = lane & 15;
    const int hi   = lane >> 4;       // 0: lanes 0-15, 1: lanes 16-31
    const int n    = (nt << 4) + l16; // this lane's hidden column (C/D layout)

    // ---- one-time async fill of the B strip into LDS (ASYNCcnt-tracked DMA) ----
    {
        const bf16_t* strip = h2hp + (size_t)nt * BSTRIP_ELEMS;
        unsigned lds_base = (unsigned)(uintptr_t)(&Bsh[0]);
#pragma unroll
        for (int c = 0; c < BSTRIP_ELEMS / 8 / 256; ++c) {       // 8 chunks of 16 B per thread
            int ci = c * 256 + threadIdx.x;
            unsigned dst = lds_base + (unsigned)ci * 16u;
            const bf16_t* src = strip + (size_t)ci * 8;
            asm volatile("global_load_async_to_lds_b128 %0, %1, off"
                         :: "v"(dst), "v"(src) : "memory");
        }
        asm volatile("s_wait_asynccnt 0" ::: "memory");
        __syncthreads();
    }

    // per-lane constants (column n)
    const float x2h_n  = x2h[n];
    const float bias_n = bias[n];
    const float gam_n  = gam[n];
    const float eps_n  = eps[n];

    const float dt    = 0.05f;
    const float theta = 0.1f;
    const float alpha = 0.951229424500714f;     // exp(-1/20)
    const float oma   = 1.0f - alpha;

    float hy[8], hz[8], u[8];
    int   mrow[8];
#pragma unroll
    for (int r = 0; r < 8; ++r) {
        hy[r] = 0.f; hz[r] = 0.f; u[r] = 0.f;
        mrow[r] = mbase + r + hi * 8;           // C/D layout: M = r + 8*hi
    }
    float spikes = 0.f;

    for (int t = 0; t < TSTEPS; ++t) {
        const bf16_t* hyrd = (t & 1) ? hyb1 : hyb0;
        bf16_t*       hywr = (t & 1) ? hyb0 : hyb1;

        v8f c = {0.f, 0.f, 0.f, 0.f, 0.f, 0.f, 0.f, 0.f};
        // A layout: all lanes M = mbase + l16; lanes<16 take K {0..15}, lanes>=16 {16..31}
        const bf16_t* ap = hyrd + (size_t)(mbase + l16) * NHID + hi * 16;
        const bf16_t* bp = Bsh + lane * 16;     // this lane's fragment, +512 elems per kstep
#pragma unroll 4
        for (int ks = 0; ks < KSTEPS; ++ks) {
            Frag16 a, b;
            a.q[0] = *(const uint4*)(ap);
            a.q[1] = *(const uint4*)(ap + 8);
            b.q[0] = *(const uint4*)(bp);       // ds_load_b128
            b.q[1] = *(const uint4*)(bp + 8);
            c = __builtin_amdgcn_wmma_f32_16x16x32_bf16(
                    false, a.v, false, b.v, (short)0, c, false, false);
            ap += 32;            // next 32 K of this hy row
            bp += 32 * 16;       // next LDS fragment
        }

        // elementwise oscillator + spike + LI update (owned patch, all in VGPRs)
#pragma unroll
        for (int r = 0; r < 8; ++r) {
            float xt  = x[mrow[r] * TSTEPS + t];            // x is (B, T, 1)
            float pre = fast_tanh(c[r] + xt * x2h_n + bias_n);
            hz[r] += dt * (pre - gam_n * hy[r] - eps_n * hz[r]);
            hy[r] += dt * hz[r];
            float s = (hy[r] - theta > 0.f) ? 1.f : 0.f;
            u[r] = u[r] * alpha + s * oma;
            spikes += s;
            hywr[(size_t)mrow[r] * NHID + n] = (bf16_t)hy[r]; // publish bf16 hy for next GEMM
        }

        __threadfence();                    // release hy(t) before the barrier
        grid_sync(cnt, gen, MAIN_BLOCKS);   // hy(t) visible before anyone reads it at t+1
    }

    // write u patch, accumulate spike total
#pragma unroll
    for (int r = 0; r < 8; ++r)
        out[(size_t)mrow[r] * NHID + n] = u[r];
    atomicAdd(tot, spikes);
    grid_sync(cnt, gen, MAIN_BLOCKS);
    if (blockIdx.x == 0 && threadIdx.x == 0) {
        volatile float* vt = tot;
        out[NBAT * NHID] = (*vt) / (float)((long long)NBAT * TSTEPS * NHID);
    }
}

extern "C" void kernel_launch(void* const* d_in, const int* in_sizes, int n_in,
                              void* d_out, int out_size, void* d_ws, size_t ws_size,
                              hipStream_t stream) {
    const float* x    = (const float*)d_in[0];   // (128, 500, 1)
    const float* h2h  = (const float*)d_in[1];   // (1024, 1024)
    const float* x2h  = (const float*)d_in[2];   // (1, 1024)
    const float* bias = (const float*)d_in[3];   // (1024,)
    const float* gam  = (const float*)d_in[4];   // (1024,)
    const float* eps  = (const float*)d_in[5];   // (1024,)

    char* ws = (char*)d_ws;
    bf16_t*   h2hp = (bf16_t*)(ws + H2HP_OFF);
    bf16_t*   hyb0 = (bf16_t*)(ws + HYB0_OFF);
    bf16_t*   hyb1 = (bf16_t*)(ws + HYB1_OFF);
    float*    tot  = (float*)(ws + TOT_OFF);
    unsigned* cnt  = (unsigned*)(ws + CNT_OFF);
    unsigned* gen  = (unsigned*)(ws + GEN_OFF);
    float*    out  = (float*)d_out;

    const int zwords = ZERO_BYTES / 4;
    hipLaunchKernelGGL(esn_init_kernel, dim3((zwords + 255) / 256), dim3(256), 0, stream,
                       (unsigned*)(ws + HYB0_OFF), zwords);
    hipLaunchKernelGGL(esn_pack_kernel, dim3(NHID * NHID / 256), dim3(256), 0, stream,
                       h2h, h2hp);
    hipLaunchKernelGGL(esn_scan_kernel, dim3(MAIN_BLOCKS), dim3(256), 0, stream,
                       x, x2h, bias, gam, eps, h2hp, hyb0, hyb1, tot, cnt, gen, out);
}
